// GNNUserBehaviorPredictor_69887707840665
// MI455X (gfx1250) — matrix operand order repaired and checked by reading, not compiled, not run on hardware
//
#include <hip/hip_runtime.h>
#include <hip/hip_bf16.h>
#include <math.h>

typedef __attribute__((ext_vector_type(16))) _Float16 v16h;
typedef __attribute__((ext_vector_type(8)))  float    v8f;
typedef __attribute__((ext_vector_type(4)))  float    v4f;

#define NEG_SLOPE 0.2f

// ---------------------------------------------------------------------------
// WMMA GEMM: C[M,N] = A[M,K] * B[K,N]   (fp32 in/out, f16 WMMA compute)
// One wave (32 threads) per 16x16 C tile. Requires M%16==0, N%16==0, K%32==0,
// and A rows 32B-aligned at the fragment bases (K multiple of 8).
// ACT: 0 = plain store, 1 = +bias then relu (bias indexed by column)
// ---------------------------------------------------------------------------
template <int ACT>
__global__ __launch_bounds__(32) void wmma_gemm_kernel(
    const float* __restrict__ A, const float* __restrict__ B,
    float* __restrict__ C, int M, int K, int N,
    const float* __restrict__ bias)
{
    const int tm   = blockIdx.x;          // M tile
    const int tn   = blockIdx.y;          // N tile
    const int lane = threadIdx.x;         // 0..31
    const int hi   = lane >> 4;           // 0 or 1
    const int l16  = lane & 15;
    const int m    = tm * 16 + l16;       // A row for this lane
    const int col  = tn * 16 + l16;       // C column for this lane

    v8f acc = {};
    for (int k0 = 0; k0 < K; k0 += 32) {
        // A fragment: ISA 16-bit A 16x32 layout.
        // lanes 0-15 : elems 0..7 -> K=k0+0..7,  elems 8..15 -> K=k0+16..23
        // lanes 16-31: elems 0..7 -> K=k0+8..15, elems 8..15 -> K=k0+24..31
        v16h a;
        {
            const v4f* ap = reinterpret_cast<const v4f*>(A + (size_t)m * K + k0 + hi * 8);
            v4f a0 = ap[0];   // K = base+0..3
            v4f a1 = ap[1];   // K = base+4..7
            v4f a2 = ap[4];   // K = base+16..19
            v4f a3 = ap[5];   // K = base+20..23
#pragma unroll
            for (int i = 0; i < 4; ++i) {
                a[i]      = (_Float16)a0[i];
                a[4 + i]  = (_Float16)a1[i];
                a[8 + i]  = (_Float16)a2[i];
                a[12 + i] = (_Float16)a3[i];
            }
        }
        // B fragment: col=l16; same K layout as A: elems 0..7 -> K=k0+hi*8+0..7,
        // elems 8..15 -> K=k0+16+hi*8+0..7 (coalesced across lanes per element)
        v16h b;
        {
            const float* bp = B + (size_t)(k0 + hi * 8) * N + col;
#pragma unroll
            for (int i = 0; i < 16; ++i) b[i] = (_Float16)bp[(size_t)(i + (i >= 8 ? 8 : 0)) * N];
        }
        acc = __builtin_amdgcn_wmma_f32_16x16x32_f16(
            /*neg_a=*/false, a, /*neg_b=*/false, b,
            /*c_mod=*/(short)0, acc, /*reuse_a=*/false, /*reuse_b=*/false);
    }

    // D layout: VGPR r -> row tm*16 + hi*8 + r, col tn*16 + l16
    const int row0 = tm * 16 + hi * 8;
    float bv = (ACT == 1) ? bias[col] : 0.f;
#pragma unroll
    for (int r = 0; r < 8; ++r) {
        float v = acc[r];
        if (ACT == 1) {
            v += bv;
            v = v > 0.f ? v : 0.f;
        }
        C[(size_t)(row0 + r) * N + col] = v;
    }
}

// ---------------------------------------------------------------------------
// alpha_s / alpha_d per (node, head)
// ---------------------------------------------------------------------------
__global__ void alpha_kernel(const float* __restrict__ h,
                             const float* __restrict__ a_src,
                             const float* __restrict__ a_dst,
                             float* __restrict__ as, float* __restrict__ ad,
                             int n_nodes, int heads, int dim)
{
    int i = blockIdx.x * blockDim.x + threadIdx.x;
    if (i >= n_nodes * heads) return;
    int n = i / heads, hh = i % heads;
    const float* hp  = h + (size_t)n * heads * dim + hh * dim;
    const float* sv  = a_src + hh * dim;
    const float* dv  = a_dst + hh * dim;
    float s = 0.f, d = 0.f;
#pragma unroll 4
    for (int j = 0; j < dim; ++j) { float v = hp[j]; s += v * sv[j]; d += v * dv[j]; }
    as[i] = s; ad[i] = d;
}

__global__ void fill_kernel(float* __restrict__ p, float v, long long n)
{
    long long i = (long long)blockIdx.x * blockDim.x + threadIdx.x;
    if (i < n) p[i] = v;
}

__device__ __forceinline__ void atomicMaxF(float* addr, float v)
{
    if (v >= 0.f) atomicMax((int*)addr, __float_as_int(v));
    else          atomicMin((unsigned int*)addr, __float_as_uint(v));
}

__device__ __forceinline__ void edge_sd(const int* __restrict__ ei, int e,
                                        int n_real, int& s, int& d)
{
    if (e < n_real) { s = ei[e]; d = ei[n_real + e]; }
    else            { s = e - n_real; d = s; }        // self loop
}

// ---------------------------------------------------------------------------
// segment max of leaky-relu(alpha_s[src]+alpha_d[dst]) over dst
// ---------------------------------------------------------------------------
__global__ void edge_max_kernel(const int* __restrict__ ei,
                                const float* __restrict__ as,
                                const float* __restrict__ ad,
                                float* __restrict__ mx,
                                int n_real, int n_nodes, int heads)
{
    int i = blockIdx.x * blockDim.x + threadIdx.x;
    int tot = (n_real + n_nodes) * heads;
    if (i >= tot) return;
    int e = i / heads, hh = i % heads;
    int s, d; edge_sd(ei, e, n_real, s, d);
    float v = as[s * heads + hh] + ad[d * heads + hh];
    v = v > 0.f ? v : NEG_SLOPE * v;
    atomicMaxF(&mx[d * heads + hh], v);
}

// ---------------------------------------------------------------------------
// e_exp = exp(e - max[dst]); denom[dst] += e_exp
// ---------------------------------------------------------------------------
__global__ void edge_exp_kernel(const int* __restrict__ ei,
                                const float* __restrict__ as,
                                const float* __restrict__ ad,
                                const float* __restrict__ mx,
                                float* __restrict__ eexp,
                                float* __restrict__ denom,
                                int n_real, int n_nodes, int heads)
{
    int i = blockIdx.x * blockDim.x + threadIdx.x;
    int tot = (n_real + n_nodes) * heads;
    if (i >= tot) return;
    int e = i / heads, hh = i % heads;
    int s, d; edge_sd(ei, e, n_real, s, d);
    float v = as[s * heads + hh] + ad[d * heads + hh];
    v = v > 0.f ? v : NEG_SLOPE * v;
    float ex = __expf(v - mx[d * heads + hh]);
    eexp[i] = ex;
    atomicAdd(&denom[d * heads + hh], ex);
}

// ---------------------------------------------------------------------------
// scatter: out[dst,h,:] += h[src,h,:] * (e_exp / denom[dst,h])
// 64 lanes per (edge,head) pair; dim==64
// ---------------------------------------------------------------------------
__global__ __launch_bounds__(256) void edge_scatter_kernel(
    const int* __restrict__ ei, const float* __restrict__ h,
    const float* __restrict__ eexp, const float* __restrict__ denom,
    float* __restrict__ outacc, int n_real, int n_nodes, int heads, int dim)
{
    int t = blockIdx.x * 256 + threadIdx.x;
    int lane = t & (dim - 1);
    int pair = t / dim;
    int tot = (n_real + n_nodes) * heads;
    if (pair >= tot) return;
    int e = pair / heads, hh = pair % heads;
    int s, d; edge_sd(ei, e, n_real, s, d);
    float alpha = eexp[pair] / (denom[d * heads + hh] + 1e-16f);
    float val = h[(size_t)s * heads * dim + hh * dim + lane] * alpha;
    atomicAdd(&outacc[(size_t)d * heads * dim + hh * dim + lane], val);
}

__global__ void bias_elu_kernel(float* __restrict__ p,
                                const float* __restrict__ bias,
                                long long total, int F)
{
    long long i = (long long)blockIdx.x * blockDim.x + threadIdx.x;
    if (i >= total) return;
    float v = p[i] + bias[(int)(i % F)];
    p[i] = v > 0.f ? v : (__expf(v) - 1.f);
}

__global__ void final_dot_kernel(const float* __restrict__ r,
                                 const float* __restrict__ w,
                                 const float* __restrict__ b,
                                 float* __restrict__ out, int n, int dim)
{
    int i = blockIdx.x * blockDim.x + threadIdx.x;
    if (i >= n) return;
    const float* rp = r + (size_t)i * dim;
    float s = b[0];
#pragma unroll 4
    for (int j = 0; j < dim; ++j) s += rp[j] * w[j];
    out[i] = s;
}

// ---------------------------------------------------------------------------
extern "C" void kernel_launch(void* const* d_in, const int* in_sizes, int n_in,
                              void* d_out, int out_size, void* d_ws, size_t ws_size,
                              hipStream_t stream)
{
    const float* x   = (const float*)d_in[0];
    const int*   ei  = (const int*)d_in[1];
    const float* W1  = (const float*)d_in[2];
    const float* as1 = (const float*)d_in[3];
    const float* ad1 = (const float*)d_in[4];
    const float* b1  = (const float*)d_in[5];
    const float* W2  = (const float*)d_in[6];
    const float* as2 = (const float*)d_in[7];
    const float* ad2 = (const float*)d_in[8];
    const float* b2  = (const float*)d_in[9];
    const float* Wp1 = (const float*)d_in[10];
    const float* bp1 = (const float*)d_in[11];
    const float* Wp2 = (const float*)d_in[12];
    const float* bp2 = (const float*)d_in[13];
    float* out = (float*)d_out;

    const int N = 50000, E = 800000, Fin = 128, H1 = 4, D = 64;
    const int F1 = H1 * D;                 // 256
    const float NEG_INF = -__builtin_huge_valf();

    // workspace layout (floats), ~120 MB total
    float* ws   = (float*)d_ws;
    float* h1   = ws;                              // N*F1  = 12.8M
    float* acc1 = h1 + (size_t)N * F1;             // N*F1  = 12.8M
    float* eexp = acc1 + (size_t)N * F1;           // (E+N)*H1 = 3.4M
    float* asb  = eexp + (size_t)(E + N) * H1;     // N*H1
    float* adb  = asb + (size_t)N * H1;            // N*H1
    float* mx   = adb + (size_t)N * H1;            // N*H1
    float* dn   = mx  + (size_t)N * H1;            // N*H1
    // layer-2 aliases inside the (now dead) h1 region
    float* h2   = h1;                              // N*D
    float* acc2 = h1 + (size_t)N * D;              // N*D
    float* rbuf = h1 + (size_t)2 * N * D;          // N*D

    const dim3 blk32(32), blk256(256);
    auto cdiv = [](long long a, long long b) { return (unsigned)((a + b - 1) / b); };

    // ===== Layer 1 =====
    wmma_gemm_kernel<0><<<dim3(N / 16, F1 / 16), blk32, 0, stream>>>(
        x, W1, h1, N, Fin, F1, nullptr);
    alpha_kernel<<<cdiv((long long)N * H1, 256), blk256, 0, stream>>>(
        h1, as1, ad1, asb, adb, N, H1, D);
    fill_kernel<<<cdiv((long long)N * H1, 256), blk256, 0, stream>>>(mx, NEG_INF, (long long)N * H1);
    fill_kernel<<<cdiv((long long)N * H1, 256), blk256, 0, stream>>>(dn, 0.f, (long long)N * H1);
    fill_kernel<<<cdiv((long long)N * F1, 256), blk256, 0, stream>>>(acc1, 0.f, (long long)N * F1);
    {
        long long tot = (long long)(E + N) * H1;
        edge_max_kernel<<<cdiv(tot, 256), blk256, 0, stream>>>(ei, asb, adb, mx, E, N, H1);
        edge_exp_kernel<<<cdiv(tot, 256), blk256, 0, stream>>>(ei, asb, adb, mx, eexp, dn, E, N, H1);
        edge_scatter_kernel<<<cdiv(tot * D, 256), blk256, 0, stream>>>(
            ei, h1, eexp, dn, acc1, E, N, H1, D);
    }
    bias_elu_kernel<<<cdiv((long long)N * F1, 256), blk256, 0, stream>>>(
        acc1, b1, (long long)N * F1, F1);

    // ===== Layer 2 =====
    wmma_gemm_kernel<0><<<dim3(N / 16, D / 16), blk32, 0, stream>>>(
        acc1, W2, h2, N, F1, D, nullptr);
    alpha_kernel<<<cdiv((long long)N, 256), blk256, 0, stream>>>(
        h2, as2, ad2, asb, adb, N, 1, D);
    fill_kernel<<<cdiv((long long)N, 256), blk256, 0, stream>>>(mx, NEG_INF, (long long)N);
    fill_kernel<<<cdiv((long long)N, 256), blk256, 0, stream>>>(dn, 0.f, (long long)N);
    fill_kernel<<<cdiv((long long)N * D, 256), blk256, 0, stream>>>(acc2, 0.f, (long long)N * D);
    {
        long long tot = (long long)(E + N);
        edge_max_kernel<<<cdiv(tot, 256), blk256, 0, stream>>>(ei, asb, adb, mx, E, N, 1);
        edge_exp_kernel<<<cdiv(tot, 256), blk256, 0, stream>>>(ei, asb, adb, mx, eexp, dn, E, N, 1);
        edge_scatter_kernel<<<cdiv(tot * D, 256), blk256, 0, stream>>>(
            ei, h2, eexp, dn, acc2, E, N, 1, D);
    }
    bias_elu_kernel<<<cdiv((long long)N * D, 256), blk256, 0, stream>>>(
        acc2, b2, (long long)N * D, D);

    // ===== MLP head =====
    wmma_gemm_kernel<1><<<dim3(N / 16, D / 16), blk32, 0, stream>>>(
        acc2, Wp1, rbuf, N, D, D, bp1);
    final_dot_kernel<<<cdiv((long long)N, 256), blk256, 0, stream>>>(
        rbuf, Wp2, bp2, out, N, D);

    (void)in_sizes; (void)n_in; (void)out_size; (void)ws_size;
}
